// APPNPNet_9414568312943
// MI455X (gfx1250) — matrix operand order repaired and checked by reading, not compiled, not run on hardware
//
#include <hip/hip_runtime.h>
#include <hip/hip_bf16.h>

typedef __attribute__((ext_vector_type(16))) __bf16 v16bf;
typedef __attribute__((ext_vector_type(8)))  __bf16 v8bf;
typedef __attribute__((ext_vector_type(8)))  float  v8f;

#define N_FEAT 512
#define HID    128
#define OUTC   64
#define KSTEPS 10
#define ALPHA  0.1f

// ---------------------------------------------------------------------------
// Pack a row-major f32 weight [Ktot x Nout] into fragment-contiguous bf16:
// frag f = ks*ntiles + t ; per lane 16 bf16 elements in WMMA B-layout order.
// One block (32 lanes) per fragment.
// ---------------------------------------------------------------------------
__global__ void pack_w_kernel(const float* __restrict__ W, __bf16* __restrict__ Wp,
                              int Nout, int ntiles) {
    const int f    = blockIdx.x;
    const int lane = threadIdx.x;             // 0..31
    const int ks   = f / ntiles;
    const int t    = f % ntiles;
    const int half = lane >> 4;
    const int l16  = lane & 15;
    const int n    = t * 16 + l16;
    const int kb   = ks * 32;
    __bf16* dst = Wp + ((size_t)f * 32 + lane) * 16;
    #pragma unroll
    for (int v = 0; v < 8; ++v) {
        const int K = kb + (v < 4 ? 2 * v : 2 * v + 8) + 8 * half;          // B 32x16 bf16 layout
        dst[2 * v]     = (__bf16)W[(size_t)K * Nout + n];
        dst[2 * v + 1] = (__bf16)W[(size_t)(K + 1) * Nout + n];
    }
}

// x f32 -> bf16, row-major (8 elements per thread, vectorizable)
__global__ void pack_x_kernel(const float* __restrict__ x, __bf16* __restrict__ xb,
                              long long total8) {
    const long long t = (long long)blockIdx.x * blockDim.x + threadIdx.x;
    if (t >= total8) return;
    const long long i = t * 8;
    #pragma unroll
    for (int j = 0; j < 8; ++j) xb[i + j] = (__bf16)x[i + j];
}

// ---------------------------------------------------------------------------
// Fused MLP: h2 = relu(x@W1+b1)@W2+b2 via bf16 WMMA (16x16x32).
// One wave32 per 16-row tile; block = 8 waves. B operands come pre-packed.
// ---------------------------------------------------------------------------
template <bool BF16X>
__global__ __launch_bounds__(256) void mlp_wmma_kernel(
    const float* __restrict__ x, const __bf16* __restrict__ xb,
    const __bf16* __restrict__ W1p, const float* __restrict__ b1,
    const __bf16* __restrict__ W2p, const float* __restrict__ b2,
    float* __restrict__ h2, int N)
{
    __shared__ __bf16 h1s[8][16][HID + 8];    // padded vs bank conflicts

    const int lane = threadIdx.x & 31;
    const int wave = threadIdx.x >> 5;
    const int half = lane >> 4;
    const int l16  = lane & 15;
    const long long m0 = ((long long)blockIdx.x * 8 + wave) * 16;
    if (m0 >= N) return;                      // uniform: EXEC stays all-ones

    long long arow = m0 + l16;
    if (arow >= N) arow = N - 1;              // N%16==0 here; safety clamp

    // -------- GEMM1: 16x512 @ 512x128, 8 N-tiles --------
    v8f acc1[8] = {};
    for (int ks = 0; ks < N_FEAT / 32; ++ks) {
        const int kb = ks * 32;
        v16bf a;
        if (BF16X) {
            const __bf16* xr = xb + arow * N_FEAT;
            // A 16x32 bf16 layout = two contiguous 8-elem runs per lane
            const v8bf lo = *(const v8bf*)(xr + kb + 8 * half);
            const v8bf hi = *(const v8bf*)(xr + kb + 16 + 8 * half);
            a = __builtin_shufflevector(lo, hi, 0, 1, 2, 3, 4, 5, 6, 7,
                                        8, 9, 10, 11, 12, 13, 14, 15);
        } else {
            const float* xr = x + arow * N_FEAT;
            #pragma unroll
            for (int v = 0; v < 8; ++v) {
                const int K = kb + (v < 4 ? 2 * v : 2 * v + 8) + 8 * half;
                a[2 * v]     = (__bf16)xr[K];
                a[2 * v + 1] = (__bf16)xr[K + 1];
            }
        }
        #pragma unroll
        for (int t = 0; t < 8; ++t) {
            const v16bf bf =
                *(const v16bf*)(W1p + (((size_t)ks * 8 + t) * 32 + lane) * 16);
            acc1[t] = __builtin_amdgcn_wmma_f32_16x16x32_bf16(
                false, a, false, bf, (short)0, acc1[t], false, false);
        }
    }

    // bias + relu, transpose through LDS (C-layout -> row-major bf16)
    #pragma unroll
    for (int t = 0; t < 8; ++t) {
        const int n = t * 16 + l16;
        const float bias = b1[n];
        #pragma unroll
        for (int r = 0; r < 8; ++r) {
            float v = acc1[t][r] + bias;
            v = v > 0.0f ? v : 0.0f;
            h1s[wave][r + 8 * half][n] = (__bf16)v;
        }
    }
    asm volatile("s_wait_dscnt 0" ::: "memory");   // same-wave LDS RAW fence

    // -------- GEMM2: 16x128 @ 128x64, 4 N-tiles --------
    v8f acc2[4] = {};
    #pragma unroll
    for (int ks = 0; ks < HID / 32; ++ks) {
        const int kb = ks * 32;
        v16bf a;
        #pragma unroll
        for (int v = 0; v < 8; ++v) {
            const int K = kb + (v < 4 ? 2 * v : 2 * v + 8) + 8 * half;
            a[2 * v]     = h1s[wave][l16][K];
            a[2 * v + 1] = h1s[wave][l16][K + 1];
        }
        #pragma unroll
        for (int t = 0; t < 4; ++t) {
            const v16bf bf =
                *(const v16bf*)(W2p + (((size_t)ks * 4 + t) * 32 + lane) * 16);
            acc2[t] = __builtin_amdgcn_wmma_f32_16x16x32_bf16(
                false, a, false, bf, (short)0, acc2[t], false, false);
        }
    }

    #pragma unroll
    for (int t = 0; t < 4; ++t) {
        const int n = t * 16 + l16;
        const float bias = b2[n];
        #pragma unroll
        for (int r = 0; r < 8; ++r) {
            const long long rr = m0 + r + 8 * half;
            if (rr < N) h2[rr * OUTC + n] = acc2[t][r] + bias;
        }
    }
}

// ---------------------------------------------------------------------------
// Edge prep: int64 -> int32 indices (N < 2^31), degree, rsqrt, per-edge norm
// ---------------------------------------------------------------------------
__global__ void edge_pack_kernel(const long long* __restrict__ ei,
                                 int* __restrict__ r32, int* __restrict__ c32,
                                 long long E) {
    const long long t = (long long)blockIdx.x * blockDim.x + threadIdx.x;
    if (t < E) {
        r32[t] = (int)ei[t];
        c32[t] = (int)ei[E + t];
    }
}

__global__ void fill1_kernel(float* __restrict__ p, int n) {
    const int t = blockIdx.x * blockDim.x + threadIdx.x;
    if (t < n) p[t] = 1.0f;                   // self-loop contribution
}

__global__ void deg_kernel(const int* __restrict__ c32,
                           float* __restrict__ deg, long long E) {
    const long long t = (long long)blockIdx.x * blockDim.x + threadIdx.x;
    if (t < E)
        __hip_atomic_fetch_add(&deg[c32[t]], 1.0f,
                               __ATOMIC_RELAXED, __HIP_MEMORY_SCOPE_AGENT);
}

__global__ void rsqrt_kernel(float* __restrict__ dis, int n) {
    const int t = blockIdx.x * blockDim.x + threadIdx.x;
    if (t < n) {
        const float d = dis[t];
        dis[t] = d > 0.0f ? __frsqrt_rn(d) : 0.0f;
    }
}

__global__ void norm_kernel(const int* __restrict__ r32, const int* __restrict__ c32,
                            const float* __restrict__ dis,
                            float* __restrict__ norm, long long E) {
    const long long t = (long long)blockIdx.x * blockDim.x + threadIdx.x;
    if (t < E) norm[t] = dis[r32[t]] * dis[c32[t]];
}

// ---------------------------------------------------------------------------
// Propagation sweeps (features 25.6 MB -> L2-resident on 192 MB L2)
// ---------------------------------------------------------------------------
__global__ void agg_init_kernel(const float* __restrict__ cur,
                                const float* __restrict__ dis,
                                float* __restrict__ agg, int N) {
    const long long t = (long long)blockIdx.x * blockDim.x + threadIdx.x;
    if (t >= (long long)N * 16) return;       // float4 granularity, 64 ch
    const long long i = t >> 4;
    const float d = dis[i];
    const float w = d * d;                    // self-loop norm
    float4 v = ((const float4*)cur)[t];
    float4 o; o.x = w * v.x; o.y = w * v.y; o.z = w * v.z; o.w = w * v.w;
    ((float4*)agg)[t] = o;
}

// 8 threads/edge, 8 channels (2 x float4) per thread; f32 HW atomics into L2.
__global__ void scatter_kernel(const int* __restrict__ r32,
                               const int* __restrict__ c32,
                               const float* __restrict__ norm,
                               const float* __restrict__ cur,
                               float* __restrict__ agg, long long E) {
    const long long t = (long long)blockIdx.x * blockDim.x + threadIdx.x;
    const long long e = t >> 3;
    if (e >= E) return;
    const int  c8 = (int)(t & 7);
    const long long r = r32[e];
    const long long c = c32[e];
    const float w = norm[e];
    const float4* src = (const float4*)cur + r * 16 + c8 * 2;
    const float4 v0 = src[0];
    const float4 v1 = src[1];
    float* dst = agg + c * OUTC + c8 * 8;
    __hip_atomic_fetch_add(dst + 0, w * v0.x, __ATOMIC_RELAXED, __HIP_MEMORY_SCOPE_AGENT);
    __hip_atomic_fetch_add(dst + 1, w * v0.y, __ATOMIC_RELAXED, __HIP_MEMORY_SCOPE_AGENT);
    __hip_atomic_fetch_add(dst + 2, w * v0.z, __ATOMIC_RELAXED, __HIP_MEMORY_SCOPE_AGENT);
    __hip_atomic_fetch_add(dst + 3, w * v0.w, __ATOMIC_RELAXED, __HIP_MEMORY_SCOPE_AGENT);
    __hip_atomic_fetch_add(dst + 4, w * v1.x, __ATOMIC_RELAXED, __HIP_MEMORY_SCOPE_AGENT);
    __hip_atomic_fetch_add(dst + 5, w * v1.y, __ATOMIC_RELAXED, __HIP_MEMORY_SCOPE_AGENT);
    __hip_atomic_fetch_add(dst + 6, w * v1.z, __ATOMIC_RELAXED, __HIP_MEMORY_SCOPE_AGENT);
    __hip_atomic_fetch_add(dst + 7, w * v1.w, __ATOMIC_RELAXED, __HIP_MEMORY_SCOPE_AGENT);
}

__global__ void combine_kernel(float* __restrict__ agg,
                               const float* __restrict__ h2, int N) {
    const long long t = (long long)blockIdx.x * blockDim.x + threadIdx.x;
    if (t >= (long long)N * 16) return;
    float4 a = ((float4*)agg)[t];
    float4 h = ((const float4*)h2)[t];
    float4 o;
    o.x = (1.0f - ALPHA) * a.x + ALPHA * h.x;
    o.y = (1.0f - ALPHA) * a.y + ALPHA * h.y;
    o.z = (1.0f - ALPHA) * a.z + ALPHA * h.z;
    o.w = (1.0f - ALPHA) * a.w + ALPHA * h.w;
    ((float4*)agg)[t] = o;
}

// ---------------------------------------------------------------------------
// log_softmax over 64 channels: one wave32 per row, 2 channels per lane
// ---------------------------------------------------------------------------
__global__ void lsm_kernel(const float* __restrict__ in,
                           float* __restrict__ out, int N) {
    const int wave = threadIdx.x >> 5;
    const int lane = threadIdx.x & 31;
    const long long i = (long long)blockIdx.x * 8 + wave;
    if (i >= N) return;
    float2 v = ((const float2*)in)[i * 32 + lane];
    float m = fmaxf(v.x, v.y);
    #pragma unroll
    for (int off = 16; off > 0; off >>= 1) m = fmaxf(m, __shfl_xor(m, off, 32));
    float s = __expf(v.x - m) + __expf(v.y - m);
    #pragma unroll
    for (int off = 16; off > 0; off >>= 1) s += __shfl_xor(s, off, 32);
    const float lg = __logf(s);
    float2 o; o.x = v.x - m - lg; o.y = v.y - m - lg;
    ((float2*)out)[i * 32 + lane] = o;
}

// ---------------------------------------------------------------------------
extern "C" void kernel_launch(void* const* d_in, const int* in_sizes, int n_in,
                              void* d_out, int out_size, void* d_ws, size_t ws_size,
                              hipStream_t stream) {
    const float*     x   = (const float*)d_in[0];
    const long long* ei  = (const long long*)d_in[1];   // int64 [2, E]
    const float*     W1  = (const float*)d_in[2];
    const float*     b1  = (const float*)d_in[3];
    const float*     W2  = (const float*)d_in[4];
    const float*     b2  = (const float*)d_in[5];
    float*           out = (float*)d_out;

    const int       N = in_sizes[0] / N_FEAT;           // 100000
    const long long E = (long long)in_sizes[1] / 2;     // 3200000

    // workspace carve-out
    char* ws = (char*)d_ws;
    size_t off = 0;
    auto carve = [&](size_t bytes) -> void* {
        void* p = ws + off;
        off = (off + bytes + 255) & ~((size_t)255);
        return p;
    };
    float*  h2   = (float*)carve((size_t)N * OUTC * sizeof(float));
    float*  bufA = (float*)carve((size_t)N * OUTC * sizeof(float));
    float*  bufB = (float*)carve((size_t)N * OUTC * sizeof(float));
    float*  dis  = (float*)carve((size_t)N * sizeof(float));
    float*  nrm  = (float*)carve((size_t)E * sizeof(float));
    int*    r32  = (int*)carve((size_t)E * sizeof(int));
    int*    c32  = (int*)carve((size_t)E * sizeof(int));
    __bf16* W1p  = (__bf16*)carve((size_t)N_FEAT * HID * sizeof(__bf16));
    __bf16* W2p  = (__bf16*)carve((size_t)HID * OUTC * sizeof(__bf16));
    const size_t base_bytes = off;
    __bf16* xb = (__bf16*)carve((size_t)N * N_FEAT * sizeof(__bf16));
    const bool use_bf16x = (off <= ws_size);            // deterministic choice

    // 0) pre-pack operands
    pack_w_kernel<<<(N_FEAT / 32) * (HID / 16), 32, 0, stream>>>(W1, W1p, HID, HID / 16);
    pack_w_kernel<<<(HID / 32) * (OUTC / 16), 32, 0, stream>>>(W2, W2p, OUTC, OUTC / 16);
    if (use_bf16x) {
        const long long total8 = ((long long)N * N_FEAT) / 8;
        pack_x_kernel<<<(int)((total8 + 255) / 256), 256, 0, stream>>>(x, xb, total8);
    }
    (void)base_bytes;

    // 1) MLP via bf16 WMMA
    if (use_bf16x)
        mlp_wmma_kernel<true><<<(N + 127) / 128, 256, 0, stream>>>(
            x, xb, W1p, b1, W2p, b2, h2, N);
    else
        mlp_wmma_kernel<false><<<(N + 127) / 128, 256, 0, stream>>>(
            x, xb, W1p, b1, W2p, b2, h2, N);

    // 2) edge prep: narrow indices, degree (self loop = 1), dis, per-edge norm
    edge_pack_kernel<<<(int)((E + 255) / 256), 256, 0, stream>>>(ei, r32, c32, E);
    fill1_kernel<<<(N + 255) / 256, 256, 0, stream>>>(dis, N);
    deg_kernel<<<(int)((E + 255) / 256), 256, 0, stream>>>(c32, dis, E);
    rsqrt_kernel<<<(N + 255) / 256, 256, 0, stream>>>(dis, N);
    norm_kernel<<<(int)((E + 255) / 256), 256, 0, stream>>>(r32, c32, dis, nrm, E);

    // 3) K propagation sweeps (ping-pong bufA/bufB, h2 constant)
    const int nvec_blocks = (int)(((long long)N * 16 + 255) / 256);
    const int edge_blocks = (int)((E * 8 + 255) / 256);
    float* cur = h2;
    float* nxt = bufA;
    for (int k = 0; k < KSTEPS; ++k) {
        agg_init_kernel<<<nvec_blocks, 256, 0, stream>>>(cur, dis, nxt, N);
        scatter_kernel<<<edge_blocks, 256, 0, stream>>>(r32, c32, nrm, cur, nxt, E);
        combine_kernel<<<nvec_blocks, 256, 0, stream>>>(nxt, h2, N);
        cur = nxt;
        nxt = (cur == bufA) ? bufB : bufA;
    }

    // 4) log_softmax -> d_out
    lsm_kernel<<<(N + 7) / 8, 256, 0, stream>>>(cur, out, N);
}
